// APPNP_59846074302983
// MI455X (gfx1250) — matrix operand order repaired and checked by reading, not compiled, not run on hardware
//
#include <hip/hip_runtime.h>
#include <hip/hip_bf16.h>

// ---------------------------------------------------------------------------
// APPNP for MI455X (gfx1250):
//   h   = relu(x @ W1^T + b1) @ W2^T + b2            (WMMA bf16, f32 accum)
//   z_k = (1-a) * A_hat z_{k-1} + a * h, 10 steps    (L2-resident f32 atomics)
//
// Roofline: MLP = 26 GFLOP / 205 MB x-read (~10 us floor each); propagation
// dominates at ~4 GB of gather+atomic traffic over 10 steps (~200 us at
// 23.3 TB/s, z_next tiles stay in the 192 MB L2). The GEMM is structured so
// every B operand is one 32B contiguous chunk per lane (2x global_load_b128
// feeding v_wmma_f32_16x16x32_bf16), software-pipelined so loads for the next
// fragment/K-step are in flight while the current WMMA chain executes.
// ---------------------------------------------------------------------------

#define N_NODES 100000
#define N_EDGES 3200000
#define F_IN    512
#define F_HID   256
#define F_OUT   16
#define K_STEPS 10
#define ALPHA   0.1f

typedef __attribute__((ext_vector_type(16))) __bf16 v16bf;
typedef __attribute__((ext_vector_type(8)))  float  v8f;

static __device__ __forceinline__ __bf16 f2bf(float f) {
  // round-to-nearest-even fp32 -> bf16
  unsigned u = __builtin_bit_cast(unsigned, f);
  unsigned r = u + 0x7FFFu + ((u >> 16) & 1u);
  unsigned short h = (unsigned short)(r >> 16);
  return __builtin_bit_cast(__bf16, h);
}

// ---------------------------------------------------------------------------
// Weight pre-pack: fragment-major bf16 so each lane's B operand is one
// contiguous 32B chunk (two global_load_b128 per v_wmma operand).
// B layout (16-bit, 32x16): lane L -> column n = L&15, element e -> K = (e>>3)*16 + (L>>4)*8 + (e&7)
// ---------------------------------------------------------------------------
__global__ void pack_w1_kernel(const float* __restrict__ w1,
                               unsigned short* __restrict__ w1p) {
  int idx = blockIdx.x * blockDim.x + threadIdx.x;   // 16*16*32*16 = 131072
  if (idx >= 16 * 16 * 32 * 16) return;
  int e = idx & 15;
  int L = (idx >> 4) & 31;
  int t = (idx >> 9) & 15;                           // N-tile (16 cols of F_HID)
  int s = idx >> 13;                                 // K-step (32 of F_IN)
  int n = t * 16 + (L & 15);
  int k = s * 32 + (e >> 3) * 16 + (L >> 4) * 8 + (e & 7);
  w1p[idx] = __builtin_bit_cast(unsigned short, f2bf(w1[n * F_IN + k]));
}

__global__ void pack_w2_kernel(const float* __restrict__ w2,
                               unsigned short* __restrict__ w2p) {
  int idx = blockIdx.x * blockDim.x + threadIdx.x;   // 8*32*16 = 4096
  if (idx >= 8 * 32 * 16) return;
  int e = idx & 15;
  int L = (idx >> 4) & 31;
  int c = idx >> 9;                                  // K-step (32 of F_HID)
  int n = L & 15;
  int k = c * 32 + (e >> 3) * 16 + (L >> 4) * 8 + (e & 7);
  w2p[idx] = __builtin_bit_cast(unsigned short, f2bf(w2[n * F_HID + k]));
}

// ---------------------------------------------------------------------------
// Fused 2-layer MLP. 256 threads = 8 waves; each wave owns a 16-row M-tile.
// Layer 1: 16 K-steps x 16 N-tiles of v_wmma_f32_16x16x32_bf16 (acc in VGPRs),
//          B fragments double-buffered, A (x) loads prefetched one K-step ahead.
// ReLU+bias tile -> LDS (bf16) -> re-read as layer-2 A fragments.
// Layer 2: 8 K-steps against packed W2, +b2, store f32 h tile.
// ---------------------------------------------------------------------------
__global__ __launch_bounds__(256) void mlp_wmma_kernel(
    const float* __restrict__ x, const unsigned short* __restrict__ w1p,
    const float* __restrict__ b1, const unsigned short* __restrict__ w2p,
    const float* __restrict__ b2, float* __restrict__ h_out) {
  __shared__ unsigned short lds_h1[8 * 16 * F_HID];  // 64 KB: 8 waves x 16x256 bf16

  const int wave  = threadIdx.x >> 5;
  const int lane  = threadIdx.x & 31;
  const int lanlo = lane & 15;
  const int half  = lane >> 4;
  const int m0    = (blockIdx.x * 8 + wave) * 16;

  const v16bf* w1f = reinterpret_cast<const v16bf*>(w1p);
  const v16bf* w2f = reinterpret_cast<const v16bf*>(w2p);

  const v8f zero = {};
  v8f acc[16];
#pragma unroll
  for (int t = 0; t < 16; ++t) acc[t] = zero;

  // A fragments: lane L holds row m = lanlo; element e -> K = (e>>3)*16 +
  // half*8 + (e&7): two contiguous 8-float fp32 runs per fragment (b128 loads),
  // prefetched one K-step ahead, converted to bf16 in VALU under the WMMAs.
  int mrow = m0 + lanlo;
  if (mrow > N_NODES - 1) mrow = N_NODES - 1;        // clamp tail (stores guarded)
  const float* xrow = x + (long)mrow * F_IN;

  float afl[16];                                     // fp32 staging for A fragment
#pragma unroll
  for (int j = 0; j < 8; ++j) {                      // prefetch K-step 0
    afl[j]     = xrow[half * 8 + j];
    afl[j + 8] = xrow[16 + half * 8 + j];
  }

  for (int s = 0; s < 16; ++s) {
    v16bf aF;
#pragma unroll
    for (int j = 0; j < 16; ++j) aF[j] = f2bf(afl[j]);

    if (s < 15) {                                    // prefetch next K-step of x
      const int k0 = (s + 1) * 32;
#pragma unroll
      for (int j = 0; j < 8; ++j) {
        afl[j]     = xrow[k0 + half * 8 + j];
        afl[j + 8] = xrow[k0 + 16 + half * 8 + j];
      }
    }

    // double-buffered B fragments: load t+1 before issuing WMMA t
    v16bf bF = w1f[(s * 16) * 32 + lane];
#pragma unroll
    for (int t = 0; t < 16; ++t) {
      v16bf bN = (t < 15) ? w1f[(s * 16 + t + 1) * 32 + lane] : bF;
      acc[t] = __builtin_amdgcn_wmma_f32_16x16x32_bf16(
          false, aF, false, bF, (short)0, acc[t], false, false);
      bF = bN;
    }
  }

  // Layer-1 epilogue: +b1, ReLU, bf16 -> LDS row-major (C layout: M=r+half*8, N=lanlo)
  unsigned short* my = lds_h1 + wave * (16 * F_HID);
#pragma unroll
  for (int t = 0; t < 16; ++t) {
    float bb = b1[t * 16 + lanlo];
#pragma unroll
    for (int r = 0; r < 8; ++r) {
      float v = acc[t][r] + bb;
      v = v > 0.0f ? v : 0.0f;
      my[(r + half * 8) * F_HID + t * 16 + lanlo] =
          __builtin_bit_cast(unsigned short, f2bf(v));
    }
  }
  __syncthreads();

  // Layer 2: [16x256] @ [256x16] in 8 WMMA steps (A from LDS, B double-buffered).
  v8f acc2 = zero;
  v16bf b2F = w2f[lane];
#pragma unroll
  for (int c = 0; c < 8; ++c) {
    v16bf aF;
#pragma unroll
    for (int e = 0; e < 16; ++e) {
      int col = c * 32 + (e >> 3) * 16 + half * 8 + (e & 7);
      aF[e] = __builtin_bit_cast(__bf16, my[lanlo * F_HID + col]);
    }
    v16bf bN = (c < 7) ? w2f[(c + 1) * 32 + lane] : b2F;
    acc2 = __builtin_amdgcn_wmma_f32_16x16x32_bf16(
        false, aF, false, b2F, (short)0, acc2, false, false);
    b2F = bN;
  }

  float bb2 = b2[lanlo];
#pragma unroll
  for (int r = 0; r < 8; ++r) {
    int row = m0 + r + half * 8;
    if (row < N_NODES)
      h_out[(long)row * F_OUT + lanlo] = acc2[r] + bb2;
  }
}

// ---------------------------------------------------------------------------
// gcn_norm: deg (with self loop) -> d^-1/2 in place
// ---------------------------------------------------------------------------
__global__ void deg_init_kernel(float* __restrict__ deg) {
  int i = blockIdx.x * blockDim.x + threadIdx.x;
  if (i < N_NODES) deg[i] = 1.0f;                    // self-loop
}

__global__ void deg_edge_kernel(const int* __restrict__ col, float* __restrict__ deg) {
  int e = blockIdx.x * blockDim.x + threadIdx.x;
  if (e < N_EDGES) atomicAdd(&deg[col[e]], 1.0f);
}

__global__ void dinv_kernel(float* __restrict__ deg) {
  int i = blockIdx.x * blockDim.x + threadIdx.x;
  if (i < N_NODES) deg[i] = rsqrtf(deg[i]);          // deg >= 1 always
}

// ---------------------------------------------------------------------------
// Propagation step. init: teleport + self-loop term. edges: scatter-add.
// dst (6.4 MB) stays L2-resident -> f32 L2 atomics.
// ---------------------------------------------------------------------------
__global__ void step_init_kernel(const float* __restrict__ src,
                                 const float* __restrict__ h,
                                 const float* __restrict__ dinv,
                                 float* __restrict__ dst) {
  int idx = blockIdx.x * blockDim.x + threadIdx.x;   // N*16
  if (idx >= N_NODES * F_OUT) return;
  float d = dinv[idx >> 4];
  dst[idx] = ALPHA * h[idx] + (1.0f - ALPHA) * d * d * src[idx];
}

__global__ void step_edge_kernel(const int* __restrict__ row,
                                 const int* __restrict__ col,
                                 const float* __restrict__ dinv,
                                 const float* __restrict__ src,
                                 float* __restrict__ dst) {
  int e = blockIdx.x * blockDim.x + threadIdx.x;
  if (e >= N_EDGES) return;
  int r = row[e], c = col[e];
  float w = (1.0f - ALPHA) * dinv[r] * dinv[c];
  const float4* zs = reinterpret_cast<const float4*>(src + (long)r * F_OUT);
  float* dp = dst + (long)c * F_OUT;
#pragma unroll
  for (int q = 0; q < 4; ++q) {
    float4 v = zs[q];
    atomicAdd(dp + q * 4 + 0, w * v.x);
    atomicAdd(dp + q * 4 + 1, w * v.y);
    atomicAdd(dp + q * 4 + 2, w * v.z);
    atomicAdd(dp + q * 4 + 3, w * v.w);
  }
}

// ---------------------------------------------------------------------------
extern "C" void kernel_launch(void* const* d_in, const int* in_sizes, int n_in,
                              void* d_out, int out_size, void* d_ws, size_t ws_size,
                              hipStream_t stream) {
  const float* x  = (const float*)d_in[0];
  const float* w1 = (const float*)d_in[1];
  const float* b1 = (const float*)d_in[2];
  const float* w2 = (const float*)d_in[3];
  const float* b2 = (const float*)d_in[4];
  const int*   ei = (const int*)d_in[5];
  const int* erow = ei;                // edge_index[0] = sources
  const int* ecol = ei + N_EDGES;      // edge_index[1] = targets
  float* out = (float*)d_out;

  // workspace carve-up (256B aligned)
  char* ws = (char*)d_ws;
  unsigned short* w1p = (unsigned short*)(ws + 0);          //   262144 B
  unsigned short* w2p = (unsigned short*)(ws + 262144);     //     8192 B
  float* dinv = (float*)(ws + 270336);                      //   400000 B
  float* hbuf = (float*)(ws + 670464);                      //  6400000 B
  float* zA   = (float*)(ws + 7070464);                     //  6400000 B
  float* zB   = (float*)(ws + 13470464);                    //  6400000 B

  const int B = 256;
  // weight pre-pack (bf16 fragment-major)
  pack_w1_kernel<<<(16 * 16 * 32 * 16) / B, B, 0, stream>>>(w1, w1p);
  pack_w2_kernel<<<(8 * 32 * 16) / B, B, 0, stream>>>(w2, w2p);

  // gcn_norm
  deg_init_kernel<<<(N_NODES + B - 1) / B, B, 0, stream>>>(dinv);
  deg_edge_kernel<<<(N_EDGES + B - 1) / B, B, 0, stream>>>(ecol, dinv);
  dinv_kernel<<<(N_NODES + B - 1) / B, B, 0, stream>>>(dinv);

  // fused MLP -> h  (8 M-tiles of 16 rows per workgroup)
  mlp_wmma_kernel<<<(N_NODES + 127) / 128, B, 0, stream>>>(x, w1p, b1, w2p, b2, hbuf);

  // K-step personalized PageRank; z starts at h; last step writes d_out.
  float* zbuf[2] = {zA, zB};
  for (int k = 0; k < K_STEPS; ++k) {
    const float* src = (k == 0) ? hbuf : zbuf[(k - 1) & 1];
    float* dst = (k == K_STEPS - 1) ? out : zbuf[k & 1];
    step_init_kernel<<<(N_NODES * F_OUT) / B, B, 0, stream>>>(src, hbuf, dinv, dst);
    step_edge_kernel<<<(N_EDGES + B - 1) / B, B, 0, stream>>>(erow, ecol, dinv, src, dst);
  }
}
